// XLSTMCell_82867099009371
// MI455X (gfx1250) — matrix operand order repaired and checked by reading, not compiled, not run on hardware
//
#include <hip/hip_runtime.h>
#include <hip/hip_bf16.h>

typedef __bf16 bf16;
typedef __attribute__((ext_vector_type(2)))  bf16  bf16x2;
typedef __attribute__((ext_vector_type(4)))  bf16  bf16x4;
typedef __attribute__((ext_vector_type(8)))  bf16  bf16x8;
typedef __attribute__((ext_vector_type(16))) bf16  bf16x16;
typedef __attribute__((ext_vector_type(4)))  float f32x4;
typedef __attribute__((ext_vector_type(8)))  float f32x8;
typedef __attribute__((ext_vector_type(4)))  unsigned u32x4;
typedef __attribute__((ext_vector_type(8)))  unsigned u32x8;

#define B_ROWS   16384
#define K_DIM    2048
#define N_DIM    5120
#define H_DIM    1024
#define BM       128
#define BN       256
#define BK       32
#define LDA      40   // padded row (elems): 64B data + 16B pad = 80B -> frags 16B aligned
#define LDB      40

__device__ __forceinline__ float sigm(float x) {
    return 1.0f / (1.0f + __expf(-x));
}
__device__ __forceinline__ float tanh_fast(float x) {
    return __builtin_fmaf(2.0f, sigm(2.0f * x), -1.0f);
}

// -----------------------------------------------------------------------------
// Kernel 0: Abf[B,2048] = bf16(concat(x, h_prev))   (one-time convert)
// -----------------------------------------------------------------------------
__global__ __launch_bounds__(256) void xlstm_convert_a(
    const float* __restrict__ x, const float* __restrict__ h_prev,
    bf16* __restrict__ Abf)
{
    const size_t i  = (size_t)blockIdx.x * blockDim.x + threadIdx.x; // f32x4 groups
    const int row   = (int)(i >> 9);          // 2048/4 = 512 groups per row
    const int col4  = (int)(i & 511) * 4;
    const float* src = (col4 < H_DIM)
        ? (x      + (size_t)row * H_DIM + col4)
        : (h_prev + (size_t)row * H_DIM + (col4 - H_DIM));
    f32x4 v = *(const f32x4*)src;
    bf16x4 bv = { (bf16)v.x, (bf16)v.y, (bf16)v.z, (bf16)v.w };
    *(bf16x4*)(Abf + (size_t)row * K_DIM + col4) = bv;
}

// -----------------------------------------------------------------------------
// Kernel 1: gates[B,5H] = Abf @ W + b   (bf16 WMMA, fp32 acc)
// Block tile 128x256, 256 threads = 8 waves in 2(M) x 4(N); each wave owns a
// 64x64 tile = 4x4 WMMA tiles -> 16 v_wmma per K-step.
// A tile staged by the Tensor Data Mover (one tensor_load_to_lds per K-step,
// padding LDS rows 64B->80B in-flight); W tile converted fp32->bf16 and stored
// K-pair-packed & transposed so B fragments are contiguous.
// -----------------------------------------------------------------------------
__global__ __launch_bounds__(256) void xlstm_gates_gemm(
    const bf16*  __restrict__ Abf,     // [B, 2048] bf16
    const float* __restrict__ W,       // [2048, 5120] fp32
    const float* __restrict__ bias,    // [5120]
    float* __restrict__ gates)         // [B, 5120]
{
    __shared__ bf16 As[BM * LDA];      // row-major A tile (TDM-padded rows)
    __shared__ bf16 Bs[BN * LDB];      // transposed W tile: Bs[n][k]

    const int tid  = threadIdx.x;
    const int m0   = blockIdx.y * BM;
    const int n0   = blockIdx.x * BN;
    const int wave = tid >> 5;
    const int lane = tid & 31;
    const int wm   = wave & 1;         // M offset wm*64
    const int wn   = wave >> 1;        // N offset wn*64
    const int lrow = lane & 15;
    const int kh   = lane >> 4;        // K-half per WMMA operand layout

    f32x8 acc[4][4];
#pragma unroll
    for (int mt = 0; mt < 4; ++mt)
#pragma unroll
        for (int nt = 0; nt < 4; ++nt)
            acc[mt][nt] = (f32x8){0.f,0.f,0.f,0.f,0.f,0.f,0.f,0.f};

    // ---- TDM descriptor group 1 (loop-invariant, wave-uniform) ----
    // data_size=1 (2B), pad_enable=1, pad_interval=3 (16 DWORDs = 64B),
    // pad_amount=3 (4 DWORDs = 16B), tensor_dim0=2048, tensor_dim1=16384,
    // tile_dim0=32, tile_dim1=128, tensor_dim0_stride=2048.
    const unsigned td0 = (unsigned)K_DIM, td1 = (unsigned)B_ROWS;
    const u32x8 tg1 = {
        (1u << 16) | (1u << 20) | (3u << 22) | (3u << 25), // size/pad ctrl
        (td0 & 0xFFFFu) << 16,                             // abar=0 | td0[15:0]
        (td0 >> 16) | ((td1 & 0xFFFFu) << 16),             // td0[31:16] | td1[15:0]
        (td1 >> 16) | ((unsigned)BK << 16),                // td1[31:16] | tile_dim0
        (unsigned)BM,                                      // tile_dim1 | tile_dim2=0
        (unsigned)K_DIM,                                   // dim0_stride[31:0]
        0u,                                                // dim0_stride[47:32] | d1str lo
        0u
    };
    const unsigned lds_as = (unsigned)(size_t)(&As[0]);

    // Per-thread B staging assignment: 4 n-cols x 4 k-pairs
    const int n4 = (tid & 63) * 4;  // 0..252
    const int kb = (tid >> 6) * 2;  // 0,2,4,6

    for (int k0 = 0; k0 < K_DIM; k0 += BK) {
        // ---- stage A: one TDM DMA for the whole 128x32 bf16 tile ----
        if (wave == 0) {
            const unsigned long long ga =
                (unsigned long long)(size_t)(Abf + (size_t)m0 * K_DIM + k0);
            const u32x4 tg0 = {
                1u,                                        // count=1, user desc
                lds_as,                                    // lds_addr
                (unsigned)ga,                              // global_addr[31:0]
                (unsigned)(ga >> 32) | (2u << 30)          // ga[56:32] | type=2
            };
            asm volatile("tensor_load_to_lds %0, %1"
                         :: "s"(tg0), "s"(tg1) : "memory");
            __builtin_amdgcn_s_wait_tensorcnt(0);
        }
        // ---- stage B: convert + transpose, packing (k,k+1) pairs as b32 ----
#pragma unroll
        for (int i = 0; i < 4; ++i) {
            const int k = kb + i * 8;
            const float* wp = W + (size_t)(k0 + k) * N_DIM + n0 + n4;
            f32x4 r0 = *(const f32x4*)wp;
            f32x4 r1 = *(const f32x4*)(wp + N_DIM);
#pragma unroll
            for (int j = 0; j < 4; ++j) {
                bf16x2 p = { (bf16)r0[j], (bf16)r1[j] };
                *(bf16x2*)(&Bs[(n4 + j) * LDB + k]) = p;
            }
            if (k0 + BK < K_DIM)
                __builtin_prefetch(wp + (size_t)BK * N_DIM, 0, 1); // global_prefetch_b8
        }
        __syncthreads();

        // ---- fragments per ISA layouts ----
        // A (16x32): lane row lrow; K = kh*8+[0..7] then 16+kh*8+[0..7]
        bf16x16 afrag[4];
#pragma unroll
        for (int mt = 0; mt < 4; ++mt) {
            const bf16* ap = &As[(wm * 64 + mt * 16 + lrow) * LDA + kh * 8];
            bf16x8 lo = *(const bf16x8*)ap;
            bf16x8 hi = *(const bf16x8*)(ap + 16);
            afrag[mt] = __builtin_shufflevector(lo, hi,
                0,1,2,3,4,5,6,7,8,9,10,11,12,13,14,15);
        }
        // B (32x16): lane col lrow; K = kh*8+[0..7] then 16+kh*8+[0..7]
        bf16x16 bfrag[4];
#pragma unroll
        for (int nt = 0; nt < 4; ++nt) {
            const bf16* bp = &Bs[(wn * 64 + nt * 16 + lrow) * LDB + kh * 8];
            bf16x8 lo = *(const bf16x8*)bp;
            bf16x8 hi = *(const bf16x8*)(bp + 16);
            bfrag[nt] = __builtin_shufflevector(lo, hi,
                0,1,2,3,4,5,6,7,8,9,10,11,12,13,14,15);
        }

#pragma unroll
        for (int mt = 0; mt < 4; ++mt)
#pragma unroll
            for (int nt = 0; nt < 4; ++nt)
                acc[mt][nt] = __builtin_amdgcn_wmma_f32_16x16x32_bf16(
                    false, afrag[mt], false, bfrag[nt],
                    (short)0, acc[mt][nt], false, false);

        __syncthreads();
    }

    // ---- epilogue: + bias, store gates ----
    // C/D layout: VGPR r -> lanes 0-15: (M=r, N=lane); lanes 16-31: (M=8+r, N=lane-16)
#pragma unroll
    for (int nt = 0; nt < 4; ++nt) {
        const int n  = n0 + wn * 64 + nt * 16 + lrow;
        const float bv = bias[n];
#pragma unroll
        for (int mt = 0; mt < 4; ++mt) {
            const int mbase = m0 + wm * 64 + mt * 16 + 8 * kh;
#pragma unroll
            for (int r = 0; r < 8; ++r) {
                gates[(size_t)(mbase + r) * N_DIM + n] = acc[mt][nt][r] + bv;
            }
        }
    }
}

// -----------------------------------------------------------------------------
// Kernel 2: per-row epilogue (LayerNorm over sigmoid(o) + cell update)
// -----------------------------------------------------------------------------
__global__ __launch_bounds__(256) void xlstm_epilogue(
    const float* __restrict__ gates,    // [B, 5120] (f,i,o,c,m each 1024)
    const float* __restrict__ c_prev,   // [B, 1024]
    const float* __restrict__ gamma,
    const float* __restrict__ beta,
    const float* __restrict__ retention,
    float* __restrict__ h_new,
    float* __restrict__ c_new)
{
    __shared__ float s_sum[256];
    __shared__ float s_sq[256];

    const int row = blockIdx.x;
    const int tid = threadIdx.x;
    const int h0  = tid * 4;
    const float* g = gates + (size_t)row * N_DIM;

    f32x4 opre = *(const f32x4*)(g + 2 * H_DIM + h0);
    float osig[4];
    float lsum = 0.f, lsq = 0.f;
#pragma unroll
    for (int j = 0; j < 4; ++j) {
        float s = sigm(opre[j]);
        osig[j] = s;
        lsum += s;
        lsq  += s * s;
    }
    s_sum[tid] = lsum;
    s_sq[tid]  = lsq;
    __syncthreads();
#pragma unroll
    for (int s = 128; s > 0; s >>= 1) {
        if (tid < s) {
            s_sum[tid] += s_sum[tid + s];
            s_sq[tid]  += s_sq[tid + s];
        }
        __syncthreads();
    }
    const float inv_h = 1.0f / (float)H_DIM;
    const float mu    = s_sum[0] * inv_h;
    const float var   = s_sq[0] * inv_h - mu * mu;
    const float rstd  = __frsqrt_rn(var + 1e-5f);

    f32x4 fpre = *(const f32x4*)(g + 0 * H_DIM + h0);
    f32x4 ipre = *(const f32x4*)(g + 1 * H_DIM + h0);
    f32x4 cpre = *(const f32x4*)(g + 3 * H_DIM + h0);
    f32x4 mpre = *(const f32x4*)(g + 4 * H_DIM + h0);
    f32x4 cpv  = *(const f32x4*)(c_prev + (size_t)row * H_DIM + h0);
    f32x4 gmv  = *(const f32x4*)(gamma + h0);
    f32x4 btv  = *(const f32x4*)(beta + h0);
    f32x4 rtv  = *(const f32x4*)(retention + h0);

    f32x4 hout, cout;
#pragma unroll
    for (int j = 0; j < 4; ++j) {
        float cp = cpv[j];
        float fg = sigm(fpre[j]);
        float ig = sigm(ipre[j]);
        float cc = tanh_fast(cpre[j]);
        float mg = sigm(mpre[j]);
        float c1 = fg * cp + ig * cc;
        float r  = rtv[j];
        float c2 = c1 * r + (1.0f - r) * cp;
        float cn = mg * c2 + (1.0f - mg) * cp;
        float ln = (osig[j] - mu) * rstd * gmv[j] + btv[j];
        float oe = sigm(ln);
        hout[j] = oe * tanh_fast(cn);
        cout[j] = cn;
    }
    *(f32x4*)(h_new + (size_t)row * H_DIM + h0) = hout;
    *(f32x4*)(c_new + (size_t)row * H_DIM + h0) = cout;
}

// -----------------------------------------------------------------------------
extern "C" void kernel_launch(void* const* d_in, const int* in_sizes, int n_in,
                              void* d_out, int out_size, void* d_ws, size_t ws_size,
                              hipStream_t stream) {
    const float* x       = (const float*)d_in[0];
    const float* h_prev  = (const float*)d_in[1];
    const float* c_prev  = (const float*)d_in[2];
    const float* W       = (const float*)d_in[3];
    const float* bias    = (const float*)d_in[4];
    const float* gamma   = (const float*)d_in[5];
    const float* beta    = (const float*)d_in[6];
    const float* retn    = (const float*)d_in[7];

    // Workspace layout: [gates fp32: 335.5 MB][Abf bf16: 67.1 MB]
    float* gates = (float*)d_ws;
    bf16*  Abf   = (bf16*)((char*)d_ws + (size_t)B_ROWS * N_DIM * sizeof(float));

    float* h_new = (float*)d_out;
    float* c_new = h_new + (size_t)B_ROWS * H_DIM;

    // 0) one-time bf16 convert of concat(x, h_prev)
    const size_t groups = (size_t)B_ROWS * K_DIM / 4;
    xlstm_convert_a<<<dim3((unsigned)(groups / 256)), 256, 0, stream>>>(x, h_prev, Abf);

    // 1) WMMA GEMM -> gates
    dim3 g1(N_DIM / BN, B_ROWS / BM);   // (20, 128)
    xlstm_gates_gemm<<<g1, 256, 0, stream>>>(Abf, W, bias, gates);

    // 2) per-row epilogue
    xlstm_epilogue<<<B_ROWS, 256, 0, stream>>>(gates, c_prev, gamma, beta, retn,
                                               h_new, c_new);
}
